// GAT_80376017977663
// MI455X (gfx1250) — matrix operand order repaired and checked by reading, not compiled, not run on hardware
//
#include <hip/hip_runtime.h>
#include <hip/hip_bf16.h>

typedef __attribute__((ext_vector_type(16))) _Float16 v16h;
typedef __attribute__((ext_vector_type(8)))  _Float16 v8h;
typedef __attribute__((ext_vector_type(8)))  float    v8f;

#define LEAKY 0.2f

union HFrag { v16h v; v8h h[2]; };

// ---------------------------------------------------------------------------
// CDNA5 async global->LDS copy (ASYNCcnt-tracked, ISA 15.18.3 op 98)
// ---------------------------------------------------------------------------
__device__ __forceinline__ void async_b128(unsigned lds_off, const _Float16* g) {
  asm volatile("global_load_async_to_lds_b128 %0, %1, off"
               :: "v"(lds_off), "v"((unsigned long long)(size_t)g)
               : "memory");
}
__device__ __forceinline__ void wait_async_le6() {
  asm volatile("s_wait_asynccnt 0x6" ::: "memory");
}
__device__ __forceinline__ void wait_async_0() {
  asm volatile("s_wait_asynccnt 0x0" ::: "memory");
}

// ---------------------------------------------------------------------------
// utility kernels
// ---------------------------------------------------------------------------
__global__ void fill_f32_kernel(float* p, float v, long n) {
  long id = blockIdx.x * (long)blockDim.x + threadIdx.x;
  if (id < n) p[id] = v;
}
__global__ void fill_f16_kernel(_Float16* p, long n) {
  long id = blockIdx.x * (long)blockDim.x + threadIdx.x;
  if (id < n) p[id] = (_Float16)0.f;
}
__global__ void cvt_f16_kernel(const float* __restrict__ in, _Float16* __restrict__ out, long n) {
  long id = blockIdx.x * (long)blockDim.x + threadIdx.x;
  if (id < n) out[id] = (_Float16)in[id];
}
// out[c*ldo + r] = in[r*C + c]   (f32 -> f16 transpose)
__global__ void transpose_f32_f16_kernel(const float* __restrict__ in, _Float16* __restrict__ out,
                                         int R, int C, int ldo) {
  long id = blockIdx.x * (long)blockDim.x + threadIdx.x;
  if (id >= (long)R * C) return;
  int r = (int)(id / C), c = (int)(id % C);
  out[(long)c * ldo + r] = (_Float16)in[id];
}

// ---------------------------------------------------------------------------
// WMMA GEMM:  C[M,Ncols](f32) = A[M,K](f16) * Bt[NbRows,K](f16)^T
// Bt rows (= output columns) padded to gridDim.x*128.  K % 32 == 0, M % 16 == 0.
// Block = 128 threads = 4 waves in 2(M) x 2(N); block tile 64M x 128N,
// wave tile 32M x 64N (2 A-frags x 4 B-frags -> 8 WMMAs per K-chunk).
// K-chunks of 32 double-buffered in LDS via async global->LDS copies.
// ---------------------------------------------------------------------------
__global__ __launch_bounds__(128)
void wmma_gemm_kernel(const _Float16* __restrict__ A, const _Float16* __restrict__ Bt,
                      float* __restrict__ C, int M, int Ncols, int K) {
  __shared__ _Float16 ldsA[2][64 * 32];    // [buf][row*32 + k]
  __shared__ _Float16 ldsB[2][128 * 32];   // [buf][col*32 + k]

  const int tid  = threadIdx.x;
  const int wave = tid >> 5;
  const int lane = tid & 31;
  const int r    = lane & 15;
  const int hi   = lane >> 4;
  const int wm   = wave & 1;               // wave M position (0/1)
  const int wn   = wave >> 1;              // wave N position (0/1)
  const int m0b  = blockIdx.y * 64;
  const int n0b  = blockIdx.x * 128;

  // ---- staging addressing ------------------------------------------------
  // A tile: 64 rows x 64B; thread stages row tid/2, two 16B chunks
  const int arow  = tid >> 1;
  const int akoff = (tid & 1) * 16;                       // halves
  const _Float16* Abase = A + (size_t)(m0b + arow) * K + akoff;
  const unsigned laA = (unsigned)(size_t)&ldsA[0][arow * 32 + akoff];
  // B tile: 128 rows x 64B; thread stages its own row (4 x 16B chunks)
  const _Float16* Bbase = Bt + (size_t)(n0b + tid) * K;
  const unsigned laB = (unsigned)(size_t)&ldsB[0][tid * 32];
  const unsigned strideA = (unsigned)(64 * 32 * sizeof(_Float16));
  const unsigned strideB = (unsigned)(128 * 32 * sizeof(_Float16));

  v8f acc[2][4] = {};
  const int nk = K >> 5;

  // prologue: stage chunk 0 into buffer 0 (6 async ops / thread)
  async_b128(laA,      Abase);
  async_b128(laA + 16, Abase + 8);
  async_b128(laB,      Bbase);
  async_b128(laB + 16, Bbase + 8);
  async_b128(laB + 32, Bbase + 16);
  async_b128(laB + 48, Bbase + 24);

  for (int c = 0; c < nk; ++c) {
    const int cur = c & 1;
    if (c + 1 < nk) {                      // stage next chunk into other buffer
      const unsigned boA = (cur ^ 1) * strideA;
      const unsigned boB = (cur ^ 1) * strideB;
      const int k1 = (c + 1) << 5;
      async_b128(laA + boA,      Abase + k1);
      async_b128(laA + boA + 16, Abase + k1 + 8);
      async_b128(laB + boB,      Bbase + k1);
      async_b128(laB + boB + 16, Bbase + k1 + 8);
      async_b128(laB + boB + 32, Bbase + k1 + 16);
      async_b128(laB + boB + 48, Bbase + k1 + 24);
      wait_async_le6();                    // current buffer's 6 ops complete
    } else {
      wait_async_0();
    }
    __syncthreads();                       // whole tile visible to all waves

    // A fragments 16x32 (ISA 7.12.2): pair-interleaved K halves per lane
    HFrag af[2];
#pragma unroll
    for (int t = 0; t < 2; ++t) {
      const _Float16* ap = &ldsA[cur][(wm * 32 + t * 16 + r) * 32];
      af[t].h[0] = *(const v8h*)(ap + hi * 8);
      af[t].h[1] = *(const v8h*)(ap + 16 + hi * 8);
    }
#pragma unroll
    for (int j = 0; j < 4; ++j) {
      const _Float16* bp = &ldsB[cur][(wn * 64 + j * 16 + r) * 32];
      HFrag bf;
      bf.h[0] = *(const v8h*)(bp + hi * 8);
      bf.h[1] = *(const v8h*)(bp + 16 + hi * 8);
      acc[0][j] = __builtin_amdgcn_wmma_f32_16x16x32_f16(
          false, af[0].v, false, bf.v, (short)0, acc[0][j], false, false);
      acc[1][j] = __builtin_amdgcn_wmma_f32_16x16x32_f16(
          false, af[1].v, false, bf.v, (short)0, acc[1][j], false, false);
    }
    __syncthreads();                       // reads done before buffer reuse
  }

  // ---- epilogue: store with M / Ncols guards -----------------------------
#pragma unroll
  for (int t = 0; t < 2; ++t) {
    const int mrow = m0b + wm * 32 + t * 16;
    if (mrow < M) {                        // wave-uniform
#pragma unroll
      for (int j = 0; j < 4; ++j) {
        int col = n0b + wn * 64 + j * 16 + r;
        if (col < Ncols) {
#pragma unroll
          for (int i = 0; i < 8; ++i) {
            int row = mrow + hi * 8 + i;   // C/D layout: VGPR i -> M = i (+8 hi lanes)
            C[(size_t)row * Ncols + col] = acc[t][j][i];
          }
        }
      }
    }
  }
}

// ---------------------------------------------------------------------------
// attention helpers
// ---------------------------------------------------------------------------
__device__ __forceinline__ void atomicMaxFloat(float* addr, float val) {
  if (val >= 0.f) atomicMax((int*)addr, __float_as_int(val));
  else            atomicMin((unsigned int*)addr, __float_as_uint(val));
}

__device__ __forceinline__ void edge_sd(long id, const int* src, const int* dst,
                                        long E, int n, int doubled, int& s, int& d) {
  if (!doubled) {
    if (id < E) { s = src[id]; d = dst[id]; }
    else        { s = d = (int)(id - E); }
  } else {
    if (id < E)          { s = src[id];         d = dst[id]; }
    else if (id < 2 * E) { s = src[id - E] + n; d = dst[id - E] + n; }
    else                 { s = d = (int)(id - 2 * E); }
  }
}

// a[n,h] = dot(hlin[n, h*C : (h+1)*C], att[h, :])
__global__ void att_score_kernel(const float* __restrict__ h, const float* __restrict__ att,
                                 float* __restrict__ out, int n, int H, int C) {
  long id = blockIdx.x * (long)blockDim.x + threadIdx.x;
  if (id >= (long)n * H) return;
  int node = (int)(id / H), hd = (int)(id % H);
  const float* hp = h + (size_t)node * H * C + (size_t)hd * C;
  const float* ap = att + (size_t)hd * C;
  float s = 0.f;
  for (int c = 0; c < C; ++c) s += hp[c] * ap[c];
  out[id] = s;
}

__global__ void edge_max_kernel(const int* __restrict__ src, const int* __restrict__ dst,
                                const float* __restrict__ as, const float* __restrict__ ad,
                                float* __restrict__ emax,
                                long E, long Etot, int nNodes, int H, int doubled) {
  long id = blockIdx.x * (long)blockDim.x + threadIdx.x;
  if (id >= Etot * H) return;
  long e = id / H; int hd = (int)(id % H);
  int s, d; edge_sd(e, src, dst, E, nNodes, doubled, s, d);
  float v = as[(long)s * H + hd] + ad[(long)d * H + hd];
  v = v > 0.f ? v : LEAKY * v;
  atomicMaxFloat(&emax[(long)d * H + hd], v);
}

__global__ void edge_exp_kernel(const int* __restrict__ src, const int* __restrict__ dst,
                                const float* __restrict__ as, const float* __restrict__ ad,
                                const float* __restrict__ emax, float* __restrict__ ep,
                                float* __restrict__ esum,
                                long E, long Etot, int nNodes, int H, int doubled) {
  long id = blockIdx.x * (long)blockDim.x + threadIdx.x;
  if (id >= Etot * H) return;
  long e = id / H; int hd = (int)(id % H);
  int s, d; edge_sd(e, src, dst, E, nNodes, doubled, s, d);
  float v = as[(long)s * H + hd] + ad[(long)d * H + hd];
  v = v > 0.f ? v : LEAKY * v;
  float p = __expf(v - emax[(long)d * H + hd]);
  ep[id] = p;
  atomicAdd(&esum[(long)d * H + hd], p);
}

// one block per edge; threads stride the H*C feature dimension
__global__ void edge_scatter_kernel(const int* __restrict__ src, const int* __restrict__ dst,
                                    const float* __restrict__ ep, const float* __restrict__ esum,
                                    const float* __restrict__ hsrc, float* __restrict__ out,
                                    long E, int nNodes, int H, int C, int doubled) {
  long e = blockIdx.x;
  int s, d; edge_sd(e, src, dst, E, nNodes, doubled, s, d);
  int F = H * C;
  const float* hp = hsrc + (size_t)s * F;
  float*       op = out  + (size_t)d * F;
  for (int f = threadIdx.x; f < F; f += blockDim.x) {
    int hd = f / C;
    float alpha = ep[e * H + hd] / esum[(size_t)d * H + hd];
    atomicAdd(op + f, alpha * hp[f]);
  }
}

__global__ void elu_bias_kernel(float* __restrict__ h, _Float16* __restrict__ h16,
                                const float* __restrict__ bias, long n, int F) {
  long id = blockIdx.x * (long)blockDim.x + threadIdx.x;
  if (id >= n * F) return;
  int f = (int)(id % F);
  float v = h[id] + bias[f];
  v = v > 0.f ? v : (__expf(v) - 1.f);
  h[id] = v;
  h16[id] = (_Float16)v;
}

// ---------------------------------------------------------------------------
// cluster pooling
// ---------------------------------------------------------------------------
__global__ void cluster_colsum_kernel(const float* __restrict__ cid, const int* __restrict__ cindex,
                                      float* __restrict__ csum, int nidx, int NC) {
  __shared__ float red[256];
  int c = blockIdx.x;
  float s = 0.f;
  for (int i = threadIdx.x; i < nidx; i += blockDim.x)
    s += cid[(long)cindex[i] * NC + c];
  red[threadIdx.x] = s;
  __syncthreads();
  for (int off = 128; off > 0; off >>= 1) {
    if ((int)threadIdx.x < off) red[threadIdx.x] += red[threadIdx.x + off];
    __syncthreads();
  }
  if (threadIdx.x == 0) csum[c] = red[0];
}

// cidT16[c*KC + i] = cluster_id[cindex[i], c]  (zero-padded K)
__global__ void build_cidT_kernel(const float* __restrict__ cid, const int* __restrict__ cindex,
                                  _Float16* __restrict__ out, int nidx, int KC, int NC) {
  long id = blockIdx.x * (long)blockDim.x + threadIdx.x;
  if (id >= (long)NC * KC) return;
  int c = (int)(id / KC), i = (int)(id % KC);
  out[id] = (i < nidx) ? (_Float16)cid[(long)cindex[i] * NC + c] : (_Float16)0.f;
}

// hgT16[f*KC + i] = h[cindex[i], f]  (zero-padded K)
__global__ void build_hgT_kernel(const float* __restrict__ h, const int* __restrict__ cindex,
                                 _Float16* __restrict__ out, int nidx, int KC, int F) {
  long id = blockIdx.x * (long)blockDim.x + threadIdx.x;
  if (id >= (long)F * KC) return;
  int f = (int)(id / KC), i = (int)(id % KC);
  out[id] = (i < nidx) ? (_Float16)h[(long)cindex[i] * F + f] : (_Float16)0.f;
}

__global__ void cf_div_kernel(float* __restrict__ cf, _Float16* __restrict__ cf16,
                              const float* __restrict__ csum, int NC, int F) {
  long id = blockIdx.x * (long)blockDim.x + threadIdx.x;
  if (id >= (long)NC * F) return;
  float v = cf[id] / csum[id / F];
  cf[id] = v;
  cf16[id] = (_Float16)v;
}

__global__ void argmax_kernel(const float* __restrict__ cid, int* __restrict__ amx, int n, int NC) {
  int i = blockIdx.x * blockDim.x + threadIdx.x;
  if (i >= n) return;
  const float* row = cid + (long)i * NC;
  float best = row[0]; int bi = 0;
  for (int c = 1; c < NC; ++c) { float v = row[c]; if (v > best) { best = v; bi = c; } }
  amx[i] = bi;
}

// xx[n]   = [h16[n] | cf16[amx[n]]],  xx[n+N] = [cf16[amx[n]] | h16[n]]
__global__ void build_xx_kernel(const _Float16* __restrict__ h16, const _Float16* __restrict__ cf16,
                                const int* __restrict__ amx, _Float16* __restrict__ xx,
                                int n, int F) {
  long id = blockIdx.x * (long)blockDim.x + threadIdx.x;
  if (id >= (long)n * 2 * F) return;
  int node = (int)(id / (2 * F)), f = (int)(id % (2 * F));
  const _Float16* hrow = h16 + (long)node * F;
  const _Float16* crow = cf16 + (long)amx[node] * F;
  _Float16 top = (f < F) ? hrow[f] : crow[f - F];
  _Float16 bot = (f < F) ? crow[f] : hrow[f - F];
  xx[(long)node * 2 * F + f] = top;
  xx[((long)node + n) * 2 * F + f] = bot;
}

// out[n,c] = mean_h acc[n, h*C + c] + b2[c]
__global__ void final_kernel(const float* __restrict__ acc, const float* __restrict__ b2,
                             float* __restrict__ out, int n2, int H, int C) {
  long id = blockIdx.x * (long)blockDim.x + threadIdx.x;
  if (id >= (long)n2 * C) return;
  int node = (int)(id / C), c = (int)(id % C);
  const float* p = acc + (long)node * H * C + c;
  float s = 0.f;
  for (int hd = 0; hd < H; ++hd) s += p[hd * C];
  out[id] = s * (1.f / H) + b2[c];
}

// ---------------------------------------------------------------------------
// host orchestration
// ---------------------------------------------------------------------------
static inline int blocks_for(long total, int bs) { return (int)((total + bs - 1) / bs); }

extern "C" void kernel_launch(void* const* d_in, const int* in_sizes, int n_in,
                              void* d_out, int out_size, void* d_ws, size_t ws_size,
                              hipStream_t stream) {
  const float* x     = (const float*)d_in[0];
  const int*   eidx  = (const int*)d_in[1];
  const float* cid   = (const float*)d_in[2];
  const int*   cidx  = (const int*)d_in[3];
  const float* W1    = (const float*)d_in[4];
  const float* as1w  = (const float*)d_in[5];
  const float* ad1w  = (const float*)d_in[6];
  const float* b1    = (const float*)d_in[7];
  const float* W2    = (const float*)d_in[8];
  const float* as2w  = (const float*)d_in[9];
  const float* ad2w  = (const float*)d_in[10];
  const float* b2    = (const float*)d_in[11];
  float* out = (float*)d_out;

  const int  FIN = 512, FH1 = 512, HID_H = 8, NC = 64, OUT2 = 49, N2OUT = 392, F2 = 1024;
  const int  Nn   = in_sizes[0] / FIN;         // 20000
  const long E    = in_sizes[1] / 2;           // 320000
  const int  NIDX = in_sizes[3];               // 10000
  const int  N2   = 2 * Nn;                    // 40000
  const long Etot1 = E + Nn;                   // +self loops
  const long Etot2 = 2 * E + N2;
  const int  KC = 10016;                       // NIDX padded to 32
  const int  W2TROWS = 512;                    // 392 padded to 128
  const int* srcv = eidx;
  const int* dstv = eidx + E;

  size_t off = 0;
  auto carve = [&](size_t bytes) -> char* {
    char* p = (char*)d_ws + off;
    off += (bytes + 255) & ~(size_t)255;
    return p;
  };
  _Float16* xh     = (_Float16*)carve((size_t)Nn * FIN * 2);
  _Float16* w1t    = (_Float16*)carve((size_t)FIN * FH1 * 2);
  float*    hlin   = (float*)carve((size_t)Nn * FH1 * 4);
  float*    asc1   = (float*)carve((size_t)Nn * HID_H * 4);
  float*    adt1   = (float*)carve((size_t)Nn * HID_H * 4);
  float*    emax1  = (float*)carve((size_t)Nn * HID_H * 4);
  float*    esum1  = (float*)carve((size_t)Nn * HID_H * 4);
  float*    ep1    = (float*)carve((size_t)Etot1 * HID_H * 4);
  float*    hout   = (float*)carve((size_t)Nn * FH1 * 4);
  _Float16* h16    = (_Float16*)carve((size_t)Nn * FH1 * 2);
  _Float16* cidT   = (_Float16*)carve((size_t)NC * KC * 2);
  _Float16* hgT    = (_Float16*)carve((size_t)FH1 * KC * 2);
  float*    csum   = (float*)carve((size_t)NC * 4);
  float*    cf     = (float*)carve((size_t)NC * FH1 * 4);
  _Float16* cf16   = (_Float16*)carve((size_t)NC * FH1 * 2);
  int*      amx    = (int*)carve((size_t)Nn * 4);
  _Float16* xx     = (_Float16*)carve((size_t)N2 * F2 * 2);
  _Float16* w2t    = (_Float16*)carve((size_t)W2TROWS * F2 * 2);
  float*    h2lin  = (float*)carve((size_t)N2 * N2OUT * 4);
  float*    asc2   = (float*)carve((size_t)N2 * HID_H * 4);
  float*    adt2   = (float*)carve((size_t)N2 * HID_H * 4);
  float*    emax2  = (float*)carve((size_t)N2 * HID_H * 4);
  float*    esum2  = (float*)carve((size_t)N2 * HID_H * 4);
  float*    ep2    = (float*)carve((size_t)Etot2 * HID_H * 4);
  float*    o2acc  = (float*)carve((size_t)N2 * N2OUT * 4);

  const int BS = 256;

  // ---- layer 1 GEMM: hlin = x @ W1 -------------------------------------
  cvt_f16_kernel<<<blocks_for((long)Nn * FIN, BS), BS, 0, stream>>>(x, xh, (long)Nn * FIN);
  transpose_f32_f16_kernel<<<blocks_for((long)FIN * FH1, BS), BS, 0, stream>>>(W1, w1t, FIN, FH1, FIN);
  {
    dim3 g(FH1 / 128, (Nn + 63) / 64);
    wmma_gemm_kernel<<<g, 128, 0, stream>>>(xh, w1t, hlin, Nn, FH1, FIN);
  }

  // ---- layer 1 attention ------------------------------------------------
  att_score_kernel<<<blocks_for((long)Nn * HID_H, BS), BS, 0, stream>>>(hlin, as1w, asc1, Nn, HID_H, 64);
  att_score_kernel<<<blocks_for((long)Nn * HID_H, BS), BS, 0, stream>>>(hlin, ad1w, adt1, Nn, HID_H, 64);
  fill_f32_kernel<<<blocks_for((long)Nn * HID_H, BS), BS, 0, stream>>>(emax1, -__builtin_huge_valf(), (long)Nn * HID_H);
  fill_f32_kernel<<<blocks_for((long)Nn * HID_H, BS), BS, 0, stream>>>(esum1, 0.f, (long)Nn * HID_H);
  edge_max_kernel<<<blocks_for(Etot1 * HID_H, BS), BS, 0, stream>>>(srcv, dstv, asc1, adt1, emax1, E, Etot1, Nn, HID_H, 0);
  edge_exp_kernel<<<blocks_for(Etot1 * HID_H, BS), BS, 0, stream>>>(srcv, dstv, asc1, adt1, emax1, ep1, esum1, E, Etot1, Nn, HID_H, 0);
  fill_f32_kernel<<<blocks_for((long)Nn * FH1, BS), BS, 0, stream>>>(hout, 0.f, (long)Nn * FH1);
  edge_scatter_kernel<<<(int)Etot1, BS, 0, stream>>>(srcv, dstv, ep1, esum1, hlin, hout, E, Nn, HID_H, 64, 0);
  elu_bias_kernel<<<blocks_for((long)Nn * FH1, BS), BS, 0, stream>>>(hout, h16, b1, Nn, FH1);

  // ---- cluster pooling (WMMA GEMM: cf = cidT * hgT^T) --------------------
  cluster_colsum_kernel<<<NC, 256, 0, stream>>>(cid, cidx, csum, NIDX, NC);
  build_cidT_kernel<<<blocks_for((long)NC * KC, BS), BS, 0, stream>>>(cid, cidx, cidT, NIDX, KC, NC);
  build_hgT_kernel<<<blocks_for((long)FH1 * KC, BS), BS, 0, stream>>>(hout, cidx, hgT, NIDX, KC, FH1);
  {
    dim3 g(FH1 / 128, 1);  // M = NC = 64
    wmma_gemm_kernel<<<g, 128, 0, stream>>>(cidT, hgT, cf, NC, FH1, KC);
  }
  cf_div_kernel<<<blocks_for((long)NC * FH1, BS), BS, 0, stream>>>(cf, cf16, csum, NC, FH1);
  argmax_kernel<<<blocks_for(Nn, BS), BS, 0, stream>>>(cid, amx, Nn, NC);
  build_xx_kernel<<<blocks_for((long)Nn * F2, BS), BS, 0, stream>>>(h16, cf16, amx, xx, Nn, FH1);

  // ---- layer 2 GEMM: h2lin = xx @ W2 ------------------------------------
  fill_f16_kernel<<<blocks_for((long)W2TROWS * F2, BS), BS, 0, stream>>>(w2t, (long)W2TROWS * F2);
  transpose_f32_f16_kernel<<<blocks_for((long)F2 * N2OUT, BS), BS, 0, stream>>>(W2, w2t, F2, N2OUT, F2);
  {
    dim3 g(W2TROWS / 128, (N2 + 63) / 64);
    wmma_gemm_kernel<<<g, 128, 0, stream>>>(xx, w2t, h2lin, N2, N2OUT, F2);
  }

  // ---- layer 2 attention + head mean ------------------------------------
  att_score_kernel<<<blocks_for((long)N2 * HID_H, BS), BS, 0, stream>>>(h2lin, as2w, asc2, N2, HID_H, OUT2);
  att_score_kernel<<<blocks_for((long)N2 * HID_H, BS), BS, 0, stream>>>(h2lin, ad2w, adt2, N2, HID_H, OUT2);
  fill_f32_kernel<<<blocks_for((long)N2 * HID_H, BS), BS, 0, stream>>>(emax2, -__builtin_huge_valf(), (long)N2 * HID_H);
  fill_f32_kernel<<<blocks_for((long)N2 * HID_H, BS), BS, 0, stream>>>(esum2, 0.f, (long)N2 * HID_H);
  edge_max_kernel<<<blocks_for(Etot2 * HID_H, BS), BS, 0, stream>>>(srcv, dstv, asc2, adt2, emax2, E, Etot2, Nn, HID_H, 1);
  edge_exp_kernel<<<blocks_for(Etot2 * HID_H, BS), BS, 0, stream>>>(srcv, dstv, asc2, adt2, emax2, ep2, esum2, E, Etot2, Nn, HID_H, 1);
  fill_f32_kernel<<<blocks_for((long)N2 * N2OUT, BS), BS, 0, stream>>>(o2acc, 0.f, (long)N2 * N2OUT);
  edge_scatter_kernel<<<(int)Etot2, BS, 0, stream>>>(srcv, dstv, ep2, esum2, h2lin, o2acc, E, Nn, HID_H, OUT2, 1);
  final_kernel<<<blocks_for((long)N2 * OUT2, BS), BS, 0, stream>>>(o2acc, b2, out, N2, HID_H, OUT2);

  (void)n_in; (void)out_size; (void)ws_size;
}
